// AttentionHead_12584254177816
// MI455X (gfx1250) — compile-verified, matrix operand order checked by reading, not run on hardware
//
#include <hip/hip_runtime.h>

// ---------------------------------------------------------------------------
// Causal attention head for MI455X (gfx1250, wave32, WMMA bf16 16x16x32).
//   B=4, T=2048, C=1024, H=64
//   pass 1: convert Wq/Wk/Wv -> bf16, transposed [h][c]
//   pass 2: q,k,v = x @ W  (4 waves/block, W staged in LDS via
//           global_load_async_to_lds_b128, double-buffered on ASYNCcnt)
//   pass 3: flash attention (online softmax, causal); K/V operands register
//           double-buffered so loads hide behind the softmax chain
// ---------------------------------------------------------------------------

#define Bsz 4
#define Tn  2048
#define Cn  1024
#define Hn  64
#define NCHUNK (Cn / 32)

typedef __attribute__((ext_vector_type(16))) __bf16   v16bf;
typedef __attribute__((ext_vector_type(8)))  float    v8f;
typedef __attribute__((ext_vector_type(8)))  unsigned v8u;

__device__ __forceinline__ v8f wmma_bf16(v16bf a, v16bf b, v8f c) {
  return __builtin_amdgcn_wmma_f32_16x16x32_bf16(false, a, false, b,
                                                 (short)0, c, false, false);
}

// fp32 -> bf16 round-to-nearest-even, scalar (raw u16)
__device__ __forceinline__ unsigned short f2bf(float f) {
  unsigned u = __float_as_uint(f);
  u += 0x7FFFu + ((u >> 16) & 1u);
  return (unsigned short)(u >> 16);
}

// two fp32 -> packed 2xbf16 dword (hw cvt_pk builtin absent on this toolchain)
__device__ __forceinline__ unsigned pack2_bf16(float f0, float f1) {
#if __has_builtin(__builtin_amdgcn_cvt_pk_bf16_f32)
  return __builtin_bit_cast(unsigned, __builtin_amdgcn_cvt_pk_bf16_f32(f0, f1));
#else
  unsigned u0 = __float_as_uint(f0), u1 = __float_as_uint(f1);
  u0 += 0x7FFFu + ((u0 >> 16) & 1u);
  u1 += 0x7FFFu + ((u1 >> 16) & 1u);
  return (u0 >> 16) | (u1 & 0xFFFF0000u);
#endif
}

// K-offset pattern for 16-bit WMMA A/B operands (16x32):
// vgpr v holds K = (v<4 ? 0 : 16) + half*8 + 2*(v&3) and K+1
// => per lane: two contiguous 16-byte chunks at byte offsets half*16, 32+half*16
__device__ __forceinline__ int kpat(int v, int half) {
  return ((v & 4) ? 16 : 0) + half * 8 + 2 * (v & 3);
}

// build a 16x32 bf16 operand from a contiguous bf16 row (2x b128 loads)
__device__ __forceinline__ v16bf load_op_bf16(const unsigned short* rowp, int half) {
  v8u u;
#pragma unroll
  for (int v = 0; v < 8; ++v) u[v] = *(const unsigned*)(rowp + kpat(v, half));
  return __builtin_bit_cast(v16bf, u);
}

// build a 16x32 bf16 A operand from a contiguous fp32 row (convert + pack)
__device__ __forceinline__ v16bf cvt_op_f32(const float* rowp, int half) {
  v8u u;
#pragma unroll
  for (int v = 0; v < 8; ++v) {
    int kk = kpat(v, half);
    u[v] = pack2_bf16(rowp[kk], rowp[kk + 1]);
  }
  return __builtin_bit_cast(v16bf, u);
}

__device__ __forceinline__ v8f zero8() {
  v8f z;
#pragma unroll
  for (int e = 0; e < 8; ++e) z[e] = 0.0f;
  return z;
}

// CDNA5 async global->LDS copy (16B per lane, tracked by ASYNCcnt).
// LDS offset = low 32 bits of the flat shared-aperture address.
__device__ __forceinline__ void async_copy_b128(const void* lds_ptr, const void* gptr) {
  unsigned lds_off = (unsigned)(unsigned long long)lds_ptr;
  asm volatile("global_load_async_to_lds_b128 %0, %1, off"
               :: "v"(lds_off), "v"((unsigned long long)gptr) : "memory");
}
__device__ __forceinline__ void wait_asynccnt0() {
  asm volatile("s_wait_asynccnt 0" ::: "memory");
}

// ---------------------------------------------------------------------------
// Pass 1: W (fp32 [C][H]) -> bf16 transposed [3][H][C]
// ---------------------------------------------------------------------------
__global__ __launch_bounds__(256)
void convert_w_kernel(const float* __restrict__ Wq, const float* __restrict__ Wk,
                      const float* __restrict__ Wv, unsigned short* __restrict__ Wt) {
  int idx = blockIdx.x * blockDim.x + threadIdx.x;
  if (idx >= 3 * Cn * Hn) return;
  int mi  = idx / (Cn * Hn);
  int rem = idx - mi * (Cn * Hn);
  int c = rem / Hn, h = rem - c * Hn;
  const float* W = (mi == 0) ? Wq : (mi == 1) ? Wk : Wv;
  Wt[(mi * Hn + h) * Cn + c] = f2bf(W[c * Hn + h]);
}

// ---------------------------------------------------------------------------
// Pass 2: QKV projection. 4 waves/block, 64 rows/block. Per 32-k chunk the
// 192x32 bf16 W slice (12KB) is async-staged into LDS, double-buffered.
// ---------------------------------------------------------------------------
__global__ __launch_bounds__(128)
void proj_qkv_kernel(const float* __restrict__ x, const unsigned short* __restrict__ Wt,
                     unsigned short* __restrict__ qb, unsigned short* __restrict__ kb,
                     unsigned short* __restrict__ vt) {
  __shared__ unsigned short wbuf[2][192 * 32];   // 2 x 12KB

  const int tid  = threadIdx.x;
  const int lane = tid & 31;
  const int wv   = tid >> 5;
  const int l16  = lane & 15, half = (lane >> 4) & 1;
  const int row0 = blockIdx.x * 64 + wv * 16;

  // stage one 12KB chunk: 768 x 16B transfers, 6 per thread
  auto issue_chunk = [&](int buf, int kc) {
#pragma unroll
    for (int rep = 0; rep < 6; ++rep) {
      int i  = rep * 128 + tid;                  // 0..767
      int rr = i >> 2;                           // W-row (mi*64+h): 0..191
      int cb = i & 3;                            // 16B column block
      async_copy_b128(&wbuf[buf][rr * 32 + cb * 8],
                      Wt + (long)rr * Cn + kc + cb * 8);
    }
  };

  v8f acc[12];
#pragma unroll
  for (int t = 0; t < 12; ++t) acc[t] = zero8();

  const float* xr = x + (long)(row0 + l16) * Cn;

  issue_chunk(0, 0);
  v16bf a = cvt_op_f32(xr, half);                // A operand for chunk 0

  for (int ic = 0; ic < NCHUNK; ++ic) {
    const int b = ic & 1;
    wait_asynccnt0();                            // my chunk-ic transfers done
    __syncthreads();                             // everyone's done; buffers safe
    if (ic + 1 < NCHUNK) issue_chunk(1 - b, (ic + 1) * 32);

    // pipeline next A operand (global loads + cvt overlap the WMMAs below)
    __builtin_prefetch(xr + (ic + 2) * 32, 0, 1);
    v16bf a_next = (ic + 1 < NCHUNK) ? cvt_op_f32(xr + (ic + 1) * 32, half) : a;

    const unsigned short* wb = &wbuf[b][0];
#pragma unroll
    for (int g = 0; g < 2; ++g) {
      v16bf bt[6];
#pragma unroll
      for (int t6 = 0; t6 < 6; ++t6) {           // ds_load_b128 pairs ...
        int t = g * 6 + t6;
        int mi = t >> 2, n0 = (t & 3) * 16;
        bt[t6] = load_op_bf16(wb + (mi * Hn + n0 + l16) * 32, half);
      }
#pragma unroll
      for (int t6 = 0; t6 < 6; ++t6)             // ... then WMMA burst
        acc[g * 6 + t6] = wmma_bf16(a, bt[t6], acc[g * 6 + t6]);
    }
    a = a_next;
  }

  // C/D layout: lane l, vgpr r -> row (r + 8*(l>>4)), col (l&15)
#pragma unroll
  for (int t = 0; t < 12; ++t) {
    const int mi = t >> 2;
    const int n  = (t & 3) * 16 + l16;
#pragma unroll
    for (int r = 0; r < 8; ++r) {
      int row = row0 + r + 8 * half;
      unsigned short val = f2bf(acc[t][r]);
      if (mi == 0)      qb[(long)row * Hn + n] = val;
      else if (mi == 1) kb[(long)row * Hn + n] = val;
      else {                                     // v stored transposed [b][h][t]
        int bi = row >> 11, tt = row & (Tn - 1);
        vt[((long)bi * Hn + n) * Tn + tt] = val;
      }
    }
  }
}

// ---------------------------------------------------------------------------
// Pass 3: flash attention, one wave per 16-query tile, 32-key chunks.
// K/V B-operands register double-buffered across the softmax chain.
// ---------------------------------------------------------------------------
__global__ __launch_bounds__(32)
void flash_attn_kernel(const unsigned short* __restrict__ qb,
                       const unsigned short* __restrict__ kb,
                       const unsigned short* __restrict__ vt,
                       float* __restrict__ out) {
  __shared__ unsigned short pst[16 * 32];   // P tile staging (C -> A layout)

  const int lane = threadIdx.x;
  const int l16 = lane & 15, half = lane >> 4;
  const int tile = blockIdx.x;
  const int bi = tile >> 7;                 // 128 query tiles per batch
  const int t0 = (tile & 127) * 16;
  const float scale = 0.03125f;             // C^-0.5 = 1/32 (reference scaling)
  const float NEG = -1e30f;

  const unsigned short* qr = qb + ((long)(bi << 11) + t0 + l16) * Hn;
  v16bf aq0 = load_op_bf16(qr, half);       // Q rows, h = 0..31
  v16bf aq1 = load_op_bf16(qr + 32, half);  // Q rows, h = 32..63

  v8f oacc[4];
#pragma unroll
  for (int hc = 0; hc < 4; ++hc) oacc[hc] = zero8();

  float mrun[8], lrun[8];
#pragma unroll
  for (int r = 0; r < 8; ++r) { mrun[r] = NEG; lrun[r] = 0.0f; }

  const unsigned short* kbase = kb + (long)(bi << 11) * Hn;
  const unsigned short* vbase = vt + (long)bi * Hn * Tn;
  const int nkb = (t0 + 16 + 31) >> 5;      // causal: chunks covering <= t0+15

  // helpers to fetch a chunk's operands
  auto load_k = [&](v16bf (&bk)[4], int key0) {
#pragma unroll
    for (int st = 0; st < 2; ++st) {
      const unsigned short* kr = kbase + (long)(key0 + st * 16 + l16) * Hn;
      bk[2 * st]     = load_op_bf16(kr, half);
      bk[2 * st + 1] = load_op_bf16(kr + 32, half);
    }
  };
  auto load_v = [&](v16bf (&bv)[4], int key0) {
#pragma unroll
    for (int hc = 0; hc < 4; ++hc)
      bv[hc] = load_op_bf16(vbase + (long)(hc * 16 + l16) * Tn + key0, half);
  };

  v16bf bk[4], bv[4];
  load_k(bk, 0);
  load_v(bv, 0);

  for (int kbk = 0; kbk < nkb; ++kbk) {
    const int key0 = kbk * 32;

    // ---- S = Q K^T with current-chunk operands ----
    v8f s[2];
#pragma unroll
    for (int st = 0; st < 2; ++st) {
      v8f z = zero8();
      z = wmma_bf16(aq0, bk[2 * st], z);
      s[st] = wmma_bf16(aq1, bk[2 * st + 1], z);
    }

    // ---- issue next-chunk loads; they retire behind the softmax below ----
    const int nk0 = (kbk + 1 < nkb) ? key0 + 32 : key0;
    v16bf nbk[4], nbv[4];
    load_k(nbk, nk0);
    load_v(nbv, nk0);

    // ---- scale + causal mask (C layout: row = r+8*half, col = key) ----
#pragma unroll
    for (int st = 0; st < 2; ++st)
#pragma unroll
      for (int r = 0; r < 8; ++r) {
        int qrow = t0 + r + 8 * half;
        int key  = key0 + st * 16 + l16;
        float v  = s[st][r] * scale;
        s[st][r] = (key <= qrow) ? v : NEG;
      }

    // ---- online softmax: row reductions across 16 lanes of each half ----
#pragma unroll
    for (int r = 0; r < 8; ++r) {
      float mx = fmaxf(s[0][r], s[1][r]);
      mx = fmaxf(mx, __shfl_xor(mx, 1, 32));
      mx = fmaxf(mx, __shfl_xor(mx, 2, 32));
      mx = fmaxf(mx, __shfl_xor(mx, 4, 32));
      mx = fmaxf(mx, __shfl_xor(mx, 8, 32));
      float mnew = fmaxf(mrun[r], mx);
      float corr = __expf(mrun[r] - mnew);
      lrun[r] *= corr;
#pragma unroll
      for (int hc = 0; hc < 4; ++hc) oacc[hc][r] *= corr;
      float p0 = __expf(s[0][r] - mnew);
      float p1 = __expf(s[1][r] - mnew);
      s[0][r] = p0; s[1][r] = p1;
      float rs = p0 + p1;
      rs += __shfl_xor(rs, 1, 32);
      rs += __shfl_xor(rs, 2, 32);
      rs += __shfl_xor(rs, 4, 32);
      rs += __shfl_xor(rs, 8, 32);
      lrun[r] += rs;
      mrun[r] = mnew;
    }

    // ---- P relayout through LDS: C-layout f32 -> A-layout bf16 [16x32] ----
#pragma unroll
    for (int st = 0; st < 2; ++st)
#pragma unroll
      for (int r = 0; r < 8; ++r)
        pst[(r + 8 * half) * 32 + st * 16 + l16] = f2bf(s[st][r]);
    asm volatile("s_wait_dscnt 0" ::: "memory");   // per-wave DS ordering fence

    v16bf ap;
    {
      v8u u;
#pragma unroll
      for (int v = 0; v < 8; ++v)
        u[v] = *(const unsigned*)&pst[l16 * 32 + kpat(v, half)];
      ap = __builtin_bit_cast(v16bf, u);
    }

    // ---- O += P V (current-chunk V operands) ----
#pragma unroll
    for (int hc = 0; hc < 4; ++hc)
      oacc[hc] = wmma_bf16(ap, bv[hc], oacc[hc]);

    asm volatile("s_wait_dscnt 0" ::: "memory");   // LDS reads done before reuse

#pragma unroll
    for (int i = 0; i < 4; ++i) { bk[i] = nbk[i]; bv[i] = nbv[i]; }
  }

  // ---- epilogue: normalize and store fp32 output [B][T][H] ----
#pragma unroll
  for (int hc = 0; hc < 4; ++hc)
#pragma unroll
    for (int r = 0; r < 8; ++r) {
      int row = t0 + r + 8 * half;
      out[((long)(bi << 11) + row) * Hn + hc * 16 + l16] = oacc[hc][r] / lrun[r];
    }
}

// ---------------------------------------------------------------------------
extern "C" void kernel_launch(void* const* d_in, const int* in_sizes, int n_in,
                              void* d_out, int out_size, void* d_ws, size_t ws_size,
                              hipStream_t stream) {
  (void)in_sizes; (void)n_in; (void)out_size; (void)ws_size;
  const float* x  = (const float*)d_in[0];
  const float* Wq = (const float*)d_in[1];
  const float* Wk = (const float*)d_in[2];
  const float* Wv = (const float*)d_in[3];
  float* out = (float*)d_out;

  // workspace layout (bf16 u16): Wt[3*H*C] | q[B*T*H] | k[B*T*H] | v_t[B*H*T]
  unsigned short* Wt = (unsigned short*)d_ws;
  unsigned short* qb = Wt + 3 * Hn * Cn;
  unsigned short* kb = qb + (long)Bsz * Tn * Hn;
  unsigned short* vt = kb + (long)Bsz * Tn * Hn;

  convert_w_kernel<<<(3 * Cn * Hn + 255) / 256, 256, 0, stream>>>(Wq, Wk, Wv, Wt);
  proj_qkv_kernel<<<Bsz * Tn / 64, 128, 0, stream>>>(x, Wt, qb, kb, vt);
  flash_attn_kernel<<<Bsz * Tn / 16, 32, 0, stream>>>(qb, kb, vt, out);
}
